// MPNEncoder_18339510354321
// MI455X (gfx1250) — matrix operand order repaired and checked by reading, not compiled, not run on hardware
//
#include <hip/hip_runtime.h>
#include <hip/hip_bf16.h>

typedef __attribute__((ext_vector_type(16))) _Float16 v16h;
typedef __attribute__((ext_vector_type(8)))  _Float16 v8h;
typedef __attribute__((ext_vector_type(8)))  float    v8f;
typedef __attribute__((ext_vector_type(4)))  unsigned int v4u;

// Problem constants (match reference)
constexpr int H    = 300;      // HIDDEN
constexpr int Hp   = 320;      // HIDDEN padded (multiple of 32 = WMMA K-step)
constexpr int AF   = 133;      // ATOM_FDIM
constexpr int BF   = 147;      // BOND_FDIM
constexpr int BFp  = 160;      // BOND_FDIM padded
constexpr int KOp  = 480;      // a_input K padded: [amsg 320 | f_atoms 133 | pad 27]
constexpr int NA   = 65536;    // NUM_ATOMS
constexpr int NB   = 131072;   // NUM_BONDS
constexpr int MNB  = 6;        // MAX_NB
constexpr int NM   = 2048;     // NUM_MOLS
constexpr int NP   = 320;      // padded weight-row count (N dim), multiple of 32
constexpr int NT2  = NP / 32;  // 10 n-macro-tiles (32 wide)

// ---------------------------------------------------------------- utilities
__global__ void k_zero(v4u* __restrict__ p, long n) {
  long i = (long)blockIdx.x * blockDim.x + threadIdx.x;
  if (i < n) { v4u z = 0; p[i] = z; }
}

// Pack the three weight matrices to zero-padded f16 [NP][Kp] (row n = weight row n).
// W_o columns are permuted to match the [amsg | f_atoms] a_input layout.
__global__ void k_pack_w(const float* __restrict__ Wi, const float* __restrict__ Wh,
                         const float* __restrict__ Wo,
                         _Float16* __restrict__ Wi16, _Float16* __restrict__ Wh16,
                         _Float16* __restrict__ Wo16) {
  int i = blockIdx.x * blockDim.x + threadIdx.x;
  const int s1 = NP * BFp, s2 = NP * Hp, s3 = NP * KOp;
  if (i < s1) {
    int n = i / BFp, k = i % BFp;
    Wi16[i] = (_Float16)((n < H && k < BF) ? Wi[n * BF + k] : 0.f);
  } else if (i < s1 + s2) {
    int j = i - s1; int n = j / Hp, k = j % Hp;
    Wh16[j] = (_Float16)((n < H && k < H) ? Wh[n * H + k] : 0.f);
  } else if (i < s1 + s2 + s3) {
    int j = i - s1 - s2; int n = j / KOp, k = j % KOp;
    float v = 0.f;
    if (n < H) {
      if (k < H)                    v = Wo[n * (AF + H) + (AF + k)];   // a_message part
      else if (k >= Hp && k < Hp + AF) v = Wo[n * (AF + H) + (k - Hp)]; // f_atoms part
    }
    Wo16[j] = (_Float16)v;
  }
}

__global__ void k_pack_bonds(const float* __restrict__ fb, _Float16* __restrict__ fb16) {
  int i = blockIdx.x * blockDim.x + threadIdx.x;
  if (i < NB * BFp) {
    int b = i / BFp, k = i % BFp;
    fb16[i] = (_Float16)(k < BF ? fb[(size_t)b * BF + k] : 0.f);
  }
}

// f_atoms part of a_input: columns [Hp, Hp+AF) = f_atoms, [Hp+AF, KOp) = 0
__global__ void k_pack_atoms(const float* __restrict__ f_atoms, _Float16* __restrict__ ai16) {
  int i = blockIdx.x * blockDim.x + threadIdx.x;
  if (i >= NA * (KOp - Hp)) return;
  int a = i / (KOp - Hp), kc = i % (KOp - Hp);
  float v = kc < AF ? f_atoms[(size_t)a * AF + kc] : 0.f;
  ai16[(size_t)a * KOp + Hp + kc] = (_Float16)v;
}

// ---------------------------------------------------------------- WMMA core
// A layout (16-bit 16x32): per lane one contiguous K-run of 16 (K = hi*16+0..15), loaded as two 8-halves.
// B layout (16-bit 32x16): this code uses the same K = hi*16+0..15 run as A (not the ISA table order).
__device__ __forceinline__ v8f wmma_mac(v16h a, v16h b, v8f c) {
  return __builtin_amdgcn_wmma_f32_16x16x32_f16(false, a, false, b, (short)0, c,
                                                false, false);
}

// 2x2 macro-tile (32x32 output per wave): 4 WMMAs per 2A+2B fragment loads.
template <int KP>
__device__ __forceinline__ void wmma_2x2(const _Float16* __restrict__ a0,
                                         const _Float16* __restrict__ b0,
                                         v8f acc[4]) {
  const _Float16* a1 = a0 + (size_t)16 * KP;
  const _Float16* b1 = b0 + (size_t)16 * KP;
#pragma unroll
  for (int kk = 0; kk < KP; kk += 32) {
    union { v16h v; v8h h[2]; } A0, A1;
    A0.h[0] = *(const v8h*)(a0 + kk); A0.h[1] = *(const v8h*)(a0 + kk + 8);
    A1.h[0] = *(const v8h*)(a1 + kk); A1.h[1] = *(const v8h*)(a1 + kk + 8);
    v16h B0 = *(const v16h*)(b0 + kk);
    v16h B1 = *(const v16h*)(b1 + kk);
    acc[0] = wmma_mac(A0.v, B0, acc[0]);
    acc[1] = wmma_mac(A0.v, B1, acc[1]);
    acc[2] = wmma_mac(A1.v, B0, acc[2]);
    acc[3] = wmma_mac(A1.v, B1, acc[3]);
  }
}

struct Tile2 { int m0, n0, r, hi; bool ok; };
__device__ __forceinline__ Tile2 tile2(int mtiles2) {
  Tile2 t;
  int lane = threadIdx.x & 31;
  int wave = threadIdx.x >> 5;
  int tile = blockIdx.x * (blockDim.x >> 5) + wave;
  t.ok = tile < mtiles2 * NT2;
  int mT = tile / NT2, nT = tile % NT2;
  t.r = lane & 15; t.hi = lane >> 4;
  t.m0 = mT * 32; t.n0 = nT * 32;
  return t;
}

// inp = f_bonds @ W_i^T ; msg = relu(inp)
__global__ void k_gemm_wi(const _Float16* __restrict__ fb16, const _Float16* __restrict__ Wi16,
                          float* __restrict__ inp, _Float16* __restrict__ msg16) {
  Tile2 t = tile2(NB / 32);
  if (!t.ok) return;
  v8f acc[4] = {};
  wmma_2x2<BFp>(fb16 + (size_t)(t.m0 + t.r) * BFp + t.hi * 16,
                Wi16 + (size_t)(t.n0 + t.r) * BFp + t.hi * 16, acc);
#pragma unroll
  for (int mi = 0; mi < 2; mi++)
#pragma unroll
    for (int ni = 0; ni < 2; ni++) {
      int n = t.n0 + ni * 16 + t.r;
      int mb = t.m0 + mi * 16 + 8 * t.hi;
      v8f c = acc[mi * 2 + ni];
      if (n < H) {
#pragma unroll
        for (int j = 0; j < 8; j++) inp[(size_t)(mb + j) * H + n] = c[j];
      }
#pragma unroll
      for (int j = 0; j < 8; j++)
        msg16[(size_t)(mb + j) * Hp + n] = (_Float16)fmaxf(c[j], 0.f);
    }
}

// msg = relu(inp + delta @ W_h^T)
__global__ void k_gemm_wh(const _Float16* __restrict__ d16, const _Float16* __restrict__ Wh16,
                          const float* __restrict__ inp, _Float16* __restrict__ msg16) {
  Tile2 t = tile2(NB / 32);
  if (!t.ok) return;
  v8f acc[4] = {};
  wmma_2x2<Hp>(d16 + (size_t)(t.m0 + t.r) * Hp + t.hi * 16,
               Wh16 + (size_t)(t.n0 + t.r) * Hp + t.hi * 16, acc);
#pragma unroll
  for (int mi = 0; mi < 2; mi++)
#pragma unroll
    for (int ni = 0; ni < 2; ni++) {
      int n = t.n0 + ni * 16 + t.r;
      int mb = t.m0 + mi * 16 + 8 * t.hi;
      v8f c = acc[mi * 2 + ni];
      if (n < H) {
#pragma unroll
        for (int j = 0; j < 8; j++)
          msg16[(size_t)(mb + j) * Hp + n] =
              (_Float16)fmaxf(c[j] + inp[(size_t)(mb + j) * H + n], 0.f);
      } else {
#pragma unroll
        for (int j = 0; j < 8; j++)
          msg16[(size_t)(mb + j) * Hp + n] = (_Float16)0.f;
      }
    }
}

// atom_hiddens = relu(a_input @ W_o^T + b) scattered into per-molecule sums
__global__ void k_gemm_wo(const _Float16* __restrict__ ai16, const _Float16* __restrict__ Wo16,
                          const float* __restrict__ bias, const int* __restrict__ mol_ids,
                          float* __restrict__ molsum) {
  Tile2 t = tile2(NA / 32);
  if (!t.ok) return;
  v8f acc[4] = {};
  wmma_2x2<KOp>(ai16 + (size_t)(t.m0 + t.r) * KOp + t.hi * 16,
                Wo16 + (size_t)(t.n0 + t.r) * KOp + t.hi * 16, acc);
#pragma unroll
  for (int mi = 0; mi < 2; mi++)
#pragma unroll
    for (int ni = 0; ni < 2; ni++) {
      int n = t.n0 + ni * 16 + t.r;
      if (n >= H) continue;
      int mb = t.m0 + mi * 16 + 8 * t.hi;
      float bn = bias[n];
      v8f c = acc[mi * 2 + ni];
#pragma unroll
      for (int j = 0; j < 8; j++) {
        float v = fmaxf(c[j] + bn, 0.f);
        atomicAdd(&molsum[(size_t)mol_ids[mb + j] * H + n], v);
      }
    }
}

// ------------------------------------------------------- gather / delta / misc
// out[a][h] = sum_j msg[a2b[a][j]][h], written with row stride outStride.
__global__ void k_gather(const _Float16* __restrict__ msg16, const int* __restrict__ a2b,
                         _Float16* __restrict__ out, int outStride) {
  int tid = blockIdx.x * blockDim.x + threadIdx.x;
  if (tid >= NA * (Hp / 8)) return;
  int a = tid / (Hp / 8);
  int h0 = (tid % (Hp / 8)) * 8;
  float s[8];
#pragma unroll
  for (int e = 0; e < 8; e++) s[e] = 0.f;
#pragma unroll
  for (int j = 0; j < MNB; j++) {
    int bnd = a2b[a * MNB + j];
    v8h x = *(const v8h*)(msg16 + (size_t)bnd * Hp + h0);
#pragma unroll
    for (int e = 0; e < 8; e++) s[e] += (float)x[e];
  }
  v8h o;
#pragma unroll
  for (int e = 0; e < 8; e++) o[e] = (_Float16)s[e];
  *(v8h*)(out + (size_t)a * outStride + h0) = o;
}

__global__ void k_delta(const _Float16* __restrict__ amsg16, const _Float16* __restrict__ msg16,
                        const int* __restrict__ b2a, const int* __restrict__ b2revb,
                        _Float16* __restrict__ d16) {
  int tid = blockIdx.x * blockDim.x + threadIdx.x;
  if (tid >= NB * (Hp / 8)) return;
  int bidx = tid / (Hp / 8);
  int h0 = (tid % (Hp / 8)) * 8;
  v8h x = *(const v8h*)(amsg16 + (size_t)b2a[bidx] * Hp + h0);
  v8h y = *(const v8h*)(msg16 + (size_t)b2revb[bidx] * Hp + h0);
  v8h o;
#pragma unroll
  for (int e = 0; e < 8; e++) o[e] = (_Float16)((float)x[e] - (float)y[e]);
  *(v8h*)(d16 + (size_t)bidx * Hp + h0) = o;
}

__global__ void k_counts(const int* __restrict__ mol_ids, float* __restrict__ counts) {
  int a = blockIdx.x * blockDim.x + threadIdx.x;
  if (a < NA) atomicAdd(&counts[mol_ids[a]], 1.f);
}

__global__ void k_finalize(const float* __restrict__ molsum, const float* __restrict__ counts,
                           float* __restrict__ out) {
  int i = blockIdx.x * blockDim.x + threadIdx.x;
  if (i >= NM * H) return;
  int m = i / H;
  out[i] = molsum[i] / fmaxf(counts[m], 1.f);
}

// ---------------------------------------------------------------- launch
extern "C" void kernel_launch(void* const* d_in, const int* in_sizes, int n_in,
                              void* d_out, int out_size, void* d_ws, size_t ws_size,
                              hipStream_t stream) {
  const float* f_atoms = (const float*)d_in[0];
  const float* f_bonds = (const float*)d_in[1];
  const int*   a2b     = (const int*)d_in[2];
  const int*   b2a     = (const int*)d_in[3];
  const int*   b2revb  = (const int*)d_in[4];
  const int*   mol_ids = (const int*)d_in[5];
  const float* W_i     = (const float*)d_in[6];
  const float* W_h     = (const float*)d_in[7];
  const float* W_o_w   = (const float*)d_in[8];
  const float* W_o_b   = (const float*)d_in[9];
  float* out = (float*)d_out;

  char* base = (char*)d_ws;
  size_t off = 0;
  auto alloc = [&](size_t bytes) -> char* {
    char* p = base + off;
    off = (off + bytes + 255) & ~(size_t)255;
    return p;
  };
  float*    inp    = (float*)   alloc((size_t)NB * H * 4);     // 157 MB
  _Float16* msg16  = (_Float16*)alloc((size_t)NB * Hp * 2);    //  84 MB
  _Float16* d16    = (_Float16*)alloc((size_t)NB * Hp * 2);    //  84 MB
  _Float16* amsg16 = (_Float16*)alloc((size_t)NA * Hp * 2);    //  42 MB
  _Float16* fb16   = (_Float16*)alloc((size_t)NB * BFp * 2);   //  42 MB
  _Float16* ai16   = (_Float16*)alloc((size_t)NA * KOp * 2);   //  63 MB
  _Float16* Wi16   = (_Float16*)alloc((size_t)NP * BFp * 2);
  _Float16* Wh16   = (_Float16*)alloc((size_t)NP * Hp * 2);
  _Float16* Wo16   = (_Float16*)alloc((size_t)NP * KOp * 2);
  float*    molsum = (float*)   alloc((size_t)NM * H * 4);     // molsum+counts contiguous
  float*    counts = (float*)   alloc((size_t)NM * 4);

  const int T = 256;
  auto blocks = [](long n, int t) { return (unsigned)((n + t - 1) / t); };

  // 1) zero molecule accumulators (msg16/d16/ai16 are fully overwritten each call)
  {
    long n16 = ((size_t)NM * H * 4 + 256 + (size_t)NM * 4) / 16;
    k_zero<<<blocks(n16, T), T, 0, stream>>>((v4u*)molsum, n16);
  }
  // 2) pack weights (W_o column-permuted), bonds, and the f_atoms half of a_input
  k_pack_w<<<blocks((long)NP * (BFp + Hp + KOp), T), T, 0, stream>>>(
      W_i, W_h, W_o_w, Wi16, Wh16, Wo16);
  k_pack_bonds<<<blocks((long)NB * BFp, T), T, 0, stream>>>(f_bonds, fb16);
  k_pack_atoms<<<blocks((long)NA * (KOp - Hp), T), T, 0, stream>>>(f_atoms, ai16);

  // 3) inp = f_bonds @ W_i^T ; msg = relu(inp)
  {
    long waves = (long)(NB / 32) * NT2;
    k_gemm_wi<<<blocks(waves * 32, T), T, 0, stream>>>(fb16, Wi16, inp, msg16);
  }
  // 4) message-passing iterations
  for (int d = 0; d < 2; d++) {
    k_gather<<<blocks((long)NA * (Hp / 8), T), T, 0, stream>>>(msg16, a2b, amsg16, Hp);
    k_delta<<<blocks((long)NB * (Hp / 8), T), T, 0, stream>>>(amsg16, msg16, b2a, b2revb, d16);
    long waves = (long)(NB / 32) * NT2;
    k_gemm_wh<<<blocks(waves * 32, T), T, 0, stream>>>(d16, Wh16, inp, msg16);
  }
  // 5) final gather writes a_message directly into a_input (cols 0..319)
  k_gather<<<blocks((long)NA * (Hp / 8), T), T, 0, stream>>>(msg16, a2b, ai16, KOp);
  k_counts<<<blocks((long)NA, T), T, 0, stream>>>(mol_ids, counts);
  // 6) output GEMM fused with per-molecule sum scatter
  {
    long waves = (long)(NA / 32) * NT2;
    k_gemm_wo<<<blocks(waves * 32, T), T, 0, stream>>>(ai16, Wo16, W_o_b, mol_ids, molsum);
  }
  // 7) per-molecule mean
  k_finalize<<<blocks((long)NM * H, T), T, 0, stream>>>(molsum, counts, out);
}
